// VectorQuantizer_31696858644923
// MI455X (gfx1250) — matrix operand order repaired and checked by reading, not compiled, not run on hardware
//
#include <hip/hip_runtime.h>
#include <hip/hip_bf16.h>

typedef _Float16 v16h __attribute__((ext_vector_type(16)));
typedef _Float16 v8h  __attribute__((ext_vector_type(8)));
typedef float    v8f  __attribute__((ext_vector_type(8)));

#define ROWS   16384      // B*N = 4*4096
#define DIM    256
#define KCODES 8192
#define EPS_F  1e-12f

#define MB_ROWS      128                    // rows per block (8 waves x 16)
#define NSPLIT       4                      // N split factor (wave parallelism)
#define COLS_PER_BLK (KCODES / NSPLIT)      // 2048
#define TILES_PER_BLK (COLS_PER_BLK / 16)   // 128
#define LDS_STRIDE   264                    // halves per staged row (256 + 8 pad)

// ---------------- Kernel 1: normalize inputs rows -> f16 -----------------
__global__ __launch_bounds__(256) void vq_prep_x(const float* __restrict__ in,
                                                 _Float16* __restrict__ xh,
                                                 unsigned long long* __restrict__ key_ws) {
    __shared__ float sdata[256];
    const int r = blockIdx.x;
    const int d = threadIdx.x;
    float v = in[r * DIM + d];
    sdata[d] = v * v;
    __syncthreads();
    for (int s = 128; s > 0; s >>= 1) {
        if (d < s) sdata[d] += sdata[d + s];
        __syncthreads();
    }
    float denom = fmaxf(sqrtf(sdata[0]), EPS_F);
    xh[r * DIM + d] = (_Float16)(v / denom);
    if (d == 0) key_ws[r] = 0xFFFFFFFFFFFFFFFFull;   // argmin key init
}

// ------------- Kernel 2: codebook -> f16, ||e||^2, zero accum ------------
__global__ __launch_bounds__(256) void vq_prep_emb(const float* __restrict__ emb,
                                                   _Float16* __restrict__ eh,
                                                   float* __restrict__ en2,
                                                   float* __restrict__ accum) {
    __shared__ float sdata[256];
    const int k = blockIdx.x;
    const int d = threadIdx.x;
    float e = emb[k * DIM + d];
    eh[k * DIM + d] = (_Float16)e;
    sdata[d] = e * e;
    __syncthreads();
    for (int s = 128; s > 0; s >>= 1) {
        if (d < s) sdata[d] += sdata[d + s];
        __syncthreads();
    }
    if (d == 0) {
        en2[k] = sdata[0];
        if (k == 0) *accum = 0.0f;
    }
}

// --------- Kernel 3: WMMA GEMM (x . e^T) with fused online argmin ---------
// Block = 8 waves = 128 rows; B tiles staged in LDS (double buffered) and
// shared by all 8 waves; 4 blocks split the 8192 codes per row-slab and
// merge their argmin via sortable-u64 global_atomic_min.
__global__ __launch_bounds__(256) void vq_argmin(const _Float16* __restrict__ xh,
                                                 const _Float16* __restrict__ eh,
                                                 const float* __restrict__ en2,
                                                 unsigned long long* __restrict__ key_ws) {
    __shared__ _Float16 bbuf[2][16 * LDS_STRIDE];   // 2 x 8.25 KB

    const int tid  = threadIdx.x;
    const int wid  = tid >> 5;          // wave32
    const int lane = tid & 31;
    const int l16  = lane & 15;
    const int hi   = lane >> 4;

    const int bid    = blockIdx.x;
    const int mblk   = bid >> 2;                 // 0..127
    const int q      = bid & 3;                  // N quarter
    const int m0     = mblk * MB_ROWS + wid * 16;
    const int n_base = q * COLS_PER_BLK;

    // staging role: thread t copies one 32B chunk of the 16x256 B tile
    const int srow = tid >> 4;          // 0..15 (tile column index)
    const int schk = tid & 15;          // 0..15 (16-half chunk along K)

    // Preload all A fragments (16 rows x 256 K) once.
    // A layout (16-bit 16x32): lane l -> row m=l%16, contiguous halves [kk+16*hi..+15]
    v16h a[8];
    const _Float16* xrow = xh + (size_t)(m0 + l16) * DIM;
#pragma unroll
    for (int i = 0; i < 8; ++i) {
        const int base = i * 32 + hi * 16;
        v16h af;
        ((v8h*)&af)[0] = *(const v8h*)(xrow + base);
        ((v8h*)&af)[1] = *(const v8h*)(xrow + base + 8);
        a[i] = af;
    }

    float best[8];
    int   bidx[8];
#pragma unroll
    for (int v = 0; v < 8; ++v) { best[v] = 3.4e38f; bidx[v] = 0; }

    // stage tile 0
    const _Float16* gsrc = eh + (size_t)(n_base + srow) * DIM + schk * 16;
    v16h r = *(const v16h*)gsrc;
    *(v16h*)&bbuf[0][srow * LDS_STRIDE + schk * 16] = r;
    __syncthreads();

    for (int it = 0; it < TILES_PER_BLK; ++it) {
        const int cur = it & 1;
        if (it + 1 < TILES_PER_BLK)                 // prefetch next tile to regs
            r = *(const v16h*)(gsrc + (size_t)(it + 1) * 16 * DIM);

        // B layout (16-bit 32x16): lane l -> col n=l%16, contiguous halves K=kk+16*hi..+15
        // Issue ALL B-fragment LDS loads first so dscnt waits stagger instead of
        // serializing LDS latency into the WMMA accumulation chain.
        const _Float16* brow = &bbuf[cur][l16 * LDS_STRIDE + hi * 16];
        v16h bfr[8];
#pragma unroll
        for (int i = 0; i < 8; ++i)
            bfr[i] = *(const v16h*)(brow + i * 32);

        v8f c = {};
#pragma unroll
        for (int i = 0; i < 8; ++i)
            c = __builtin_amdgcn_wmma_f32_16x16x32_f16(
                    false, a[i], false, bfr[i], (short)0, c, false, false);

        const int n  = n_base + it * 16 + l16;
        const float e2 = en2[n];
#pragma unroll
        for (int v = 0; v < 8; ++v) {
            // distance(m,n) - ||x||^2 = ||e_n||^2 - 2*dot; strict < keeps lowest n
            float s = e2 - 2.0f * c[v];
            if (s < best[v]) { best[v] = s; bidx[v] = n; }
        }

        __syncthreads();
        if (it + 1 < TILES_PER_BLK)
            *(v16h*)&bbuf[cur ^ 1][srow * LDS_STRIDE + schk * 16] = r;
        __syncthreads();
    }

    // Sortable key: (monotonic(score) << 32) | index  -> u64 min == reference argmin
#pragma unroll
    for (int v = 0; v < 8; ++v) {
        unsigned int sb = __float_as_uint(best[v]);
        sb = (sb & 0x80000000u) ? ~sb : (sb | 0x80000000u);
        unsigned long long key =
            ((unsigned long long)sb << 32) | (unsigned int)bidx[v];
#pragma unroll
        for (int off = 1; off < 16; off <<= 1) {
            unsigned long long ok = __shfl_xor(key, off, 32);
            if (ok < key) key = ok;
        }
        if (l16 == 0)
            atomicMin(&key_ws[m0 + v + 8 * hi], key);   // global_atomic_min_u64
    }
}

// ------- Kernel 4: gather codes, straight-through output, loss partial ----
__global__ __launch_bounds__(256) void vq_output(const float* __restrict__ in,
                                                 const float* __restrict__ emb,
                                                 const unsigned long long* __restrict__ key_ws,
                                                 float* __restrict__ out_q,
                                                 float* __restrict__ out_idx_f,
                                                 float* __restrict__ accum) {
    __shared__ float sdata[256];
    const int r = blockIdx.x;
    const int d = threadIdx.x;
    float v = in[r * DIM + d];
    sdata[d] = v * v;
    __syncthreads();
    for (int s = 128; s > 0; s >>= 1) {
        if (d < s) sdata[d] += sdata[d + s];
        __syncthreads();
    }
    float denom = fmaxf(sqrtf(sdata[0]), EPS_F);
    float x = v / denom;                       // reference's normalized x (f32)
    const int k = (int)(key_ws[r] & 0xFFFFFFFFull);
    float qv = emb[k * DIM + d];
    float diff = qv - x;
    out_q[r * DIM + d] = x + diff;             // straight-through: x + (q - x)
    __syncthreads();
    sdata[d] = diff * diff;
    __syncthreads();
    for (int s = 128; s > 0; s >>= 1) {
        if (d < s) sdata[d] += sdata[d + s];
        __syncthreads();
    }
    if (d == 0) {
        atomicAdd(accum, sdata[0]);
        out_idx_f[r] = (float)k;
    }
}

// ---------------------- Kernel 5: finalize loss ---------------------------
__global__ void vq_loss_finalize(const float* __restrict__ accum,
                                 float* __restrict__ loss_out) {
    *loss_out = (*accum) * (1.0f / ((float)ROWS * (float)DIM));
}

extern "C" void kernel_launch(void* const* d_in, const int* in_sizes, int n_in,
                              void* d_out, int out_size, void* d_ws, size_t ws_size,
                              hipStream_t stream) {
    const float* inputs = (const float*)d_in[0];   // [4,4096,256] f32
    const float* emb    = (const float*)d_in[1];   // [8192,256] f32 (pre-normalized)

    float* out = (float*)d_out;
    float* out_quant = out;                         // ROWS*DIM floats
    float* out_loss  = out + (size_t)ROWS * DIM;    // 1 float
    float* out_idx_f = out_loss + 1;                // ROWS floats (indices as f32)

    // Workspace carve-up (~12.3 MB)
    char* ws = (char*)d_ws;
    _Float16* xh  = (_Float16*)(ws);                                     // 8 MB
    _Float16* eh  = (_Float16*)(ws + (size_t)ROWS * DIM * 2);            // 4 MB
    float*    en2 = (float*)(ws + (size_t)ROWS * DIM * 2
                                + (size_t)KCODES * DIM * 2);             // 32 KB
    unsigned long long* keys =
        (unsigned long long*)((char*)en2 + (size_t)KCODES * 4);          // 128 KB
    float* accum = (float*)((char*)keys + (size_t)ROWS * 8);             // 4 B

    vq_prep_x   <<<ROWS,   256, 0, stream>>>(inputs, xh, keys);
    vq_prep_emb <<<KCODES, 256, 0, stream>>>(emb, eh, en2, accum);
    vq_argmin   <<<(ROWS / MB_ROWS) * NSPLIT, 256, 0, stream>>>(xh, eh, en2, keys);
    vq_output   <<<ROWS,   256, 0, stream>>>(inputs, emb, keys, out_quant,
                                             out_idx_f, accum);
    vq_loss_finalize<<<1, 1, 0, stream>>>(accum, out_loss);
}
